// AttentionBlock_10428180594725
// MI455X (gfx1250) — matrix operand order repaired and checked by reading, not compiled, not run on hardware
//
#include <hip/hip_runtime.h>
#include <hip/hip_bf16.h>

#define D_MODEL 1024
#define N_HEADS 16
#define D_HEAD  64
#define D_FF    4096
#define WINDOW  256
#define BB      2
#define TT      2048
#define MROWS   (BB * TT)          // 4096
#define TPAD    (TT + 64)          // padded Vt row length

typedef __attribute__((ext_vector_type(16))) __bf16 v16bf;
typedef __attribute__((ext_vector_type(8)))  float  v8f;

struct Chunk32 { uint4 lo, hi; };

__device__ __forceinline__ v16bf make_frag(uint4 lo, uint4 hi) {
    Chunk32 t{lo, hi};
    return __builtin_bit_cast(v16bf, t);
}

__device__ __forceinline__ __bf16 f2bf(float f) {
    unsigned u = __builtin_bit_cast(unsigned, f);
    unsigned r = (u + 0x7FFFu + ((u >> 16) & 1u)) >> 16;
    return __builtin_bit_cast(__bf16, (unsigned short)r);
}

__device__ __forceinline__ v8f wmma_bf16(v16bf a, v16bf b, v8f c) {
    return __builtin_amdgcn_wmma_f32_16x16x32_bf16(
        false, a, false, b, (short)0, c, false, false);
}

// async copy: 16 bytes per lane, global -> LDS, tracked by ASYNCcnt
__device__ __forceinline__ void async_ld_b128(unsigned lds_addr, unsigned gl_off,
                                              const void* base) {
    asm volatile("global_load_async_to_lds_b128 %0, %1, %2"
                 :: "v"(lds_addr), "v"(gl_off), "s"(base) : "memory");
}

// ---------------- fp32 -> bf16 conversion ----------------
__global__ __launch_bounds__(256) void cvt_bf16_kernel(
    const float* __restrict__ in, __bf16* __restrict__ out, size_t n) {
    size_t stride = (size_t)gridDim.x * 256u;
    for (size_t i = (size_t)blockIdx.x * 256u + threadIdx.x; i < n; i += stride)
        out[i] = f2bf(in[i]);
}

// ---------------- RMSNorm (one row per block) ----------------
__global__ __launch_bounds__(256) void rmsnorm_kernel(
    const float* __restrict__ x, const float* __restrict__ w,
    __bf16* __restrict__ out) {
    __shared__ float red[8];
    __shared__ float s_rstd;
    const int tid = threadIdx.x;
    const float* xr = x + (size_t)blockIdx.x * D_MODEL;
    float ss = 0.f;
    for (int i = tid; i < D_MODEL; i += 256) { float v = xr[i]; ss += v * v; }
    #pragma unroll
    for (int m = 16; m; m >>= 1) ss += __shfl_xor(ss, m, 32);
    if ((tid & 31) == 0) red[tid >> 5] = ss;
    __syncthreads();
    if (tid == 0) {
        float t = 0.f;
        #pragma unroll
        for (int i = 0; i < 8; ++i) t += red[i];
        s_rstd = rsqrtf(t / (float)D_MODEL + 1e-6f);
    }
    __syncthreads();
    float rs = s_rstd;
    __bf16* orow = out + (size_t)blockIdx.x * D_MODEL;
    for (int i = tid; i < D_MODEL; i += 256) orow[i] = f2bf(xr[i] * rs * w[i]);
}

// ---------------- V transpose (+zero tail pad) ----------------
__global__ __launch_bounds__(256) void vtrans_kernel(
    const __bf16* __restrict__ qkv, __bf16* __restrict__ vT) {
    const size_t n = (size_t)BB * N_HEADS * D_HEAD * TPAD;
    size_t stride = (size_t)gridDim.x * 256u;
    for (size_t i = (size_t)blockIdx.x * 256u + threadIdx.x; i < n; i += stride) {
        int t = (int)(i % TPAD);
        size_t rest = i / TPAD;
        int d  = (int)(rest % D_HEAD);
        int bh = (int)(rest / D_HEAD);
        int h = bh % N_HEADS, b = bh / N_HEADS;
        __bf16 v = f2bf(0.f);
        if (t < TT)
            v = qkv[(size_t)(b * TT + t) * 3072 + 2048 + h * D_HEAD + d];
        vT[i] = v;
    }
}

// ---------------- generic bf16 GEMM: C[M,N] = act(A[M,K] @ W[N,K]^T) --------
// Block tile 128x128, 8 waves each 32x64 (2x4 WMMA tiles).
// Double-buffered LDS filled with global_load_async_to_lds_b128 (ASYNCcnt).
// ops: 0 = f32 store, 1 = bf16 store, 2 = sigmoid f32, 3 = bf16(silu(acc)*aux),
//      4 = f32(acc + aux)
__global__ __launch_bounds__(256) void gemm_bf16_kernel(
    const __bf16* __restrict__ A, int lda,
    const __bf16* __restrict__ W, int ldw,
    int K, void* __restrict__ Dv, int ldd,
    int op, const float* __restrict__ aux) {
    __shared__ __align__(16) __bf16 sA[2][128 * 40];
    __shared__ __align__(16) __bf16 sW[2][128 * 40];
    const int tid  = threadIdx.x;
    const int lane = tid & 31;
    const int wave = tid >> 5;
    const int m0 = blockIdx.y * 128;
    const int n0 = blockIdx.x * 128;
    const int wm = (wave >> 1) * 32;   // 0,32,64,96
    const int wn = (wave & 1) * 64;    // 0,64
    const int lr = lane & 15;
    const int hi = lane >> 4;

    // per-thread 16B-chunk coordinates: 2 chunks per 128x32 tile per thread
    int rowC[2], colC[2];
    #pragma unroll
    for (int i = 0; i < 2; ++i) {
        int c = tid + i * 256;           // 0..511
        rowC[i] = c >> 2;
        colC[i] = (c & 3) << 3;
    }

    auto issue = [&](int buf, int k0) {
        #pragma unroll
        for (int i = 0; i < 2; ++i) {
            unsigned lds = (unsigned)(size_t)&sA[buf][rowC[i] * 40 + colC[i]];
            unsigned go  = (unsigned)(((size_t)(m0 + rowC[i]) * lda + k0 + colC[i]) * 2);
            async_ld_b128(lds, go, A);
        }
        #pragma unroll
        for (int i = 0; i < 2; ++i) {
            unsigned lds = (unsigned)(size_t)&sW[buf][rowC[i] * 40 + colC[i]];
            unsigned go  = (unsigned)(((size_t)(n0 + rowC[i]) * ldw + k0 + colC[i]) * 2);
            async_ld_b128(lds, go, W);
        }
    };

    v8f acc[2][4] = {};
    issue(0, 0);

    for (int k0 = 0; k0 < K; k0 += 32) {
        const int cur = (k0 >> 5) & 1;
        const bool more = (k0 + 32) < K;
        if (more) issue(cur ^ 1, k0 + 32);   // overlap with compute on cur
        if (more) asm volatile("s_wait_asynccnt 0x4" ::: "memory");
        else      asm volatile("s_wait_asynccnt 0x0" ::: "memory");
        __syncthreads();                      // all waves' cur tile resident

        const int akb = hi ? 8 : 0;
        const int bkb = hi ? 8 : 0;
        v16bf af[2], bfm[4];
        #pragma unroll
        for (int s = 0; s < 2; ++s) {
            const __bf16* pa = &sA[cur][(wm + s * 16 + lr) * 40 + akb];
            af[s] = make_frag(*(const uint4*)pa, *(const uint4*)(pa + 16));
        }
        #pragma unroll
        for (int j = 0; j < 4; ++j) {
            const __bf16* pb = &sW[cur][(wn + j * 16 + lr) * 40 + bkb];
            bfm[j] = make_frag(*(const uint4*)pb, *(const uint4*)(pb + 16));
        }
        #pragma unroll
        for (int i = 0; i < 2; ++i)
            #pragma unroll
            for (int j = 0; j < 4; ++j)
                acc[i][j] = wmma_bf16(af[i], bfm[j], acc[i][j]);
        __syncthreads();                      // cur fully consumed before reuse
    }

    #pragma unroll
    for (int i = 0; i < 2; ++i) {
        #pragma unroll
        for (int j = 0; j < 4; ++j) {
            #pragma unroll
            for (int r = 0; r < 8; ++r) {
                int row = m0 + wm + i * 16 + r + hi * 8;
                int col = n0 + wn + j * 16 + lr;
                size_t idx = (size_t)row * ldd + col;
                float v = acc[i][j][r];
                if (op == 0) {
                    ((float*)Dv)[idx] = v;
                } else if (op == 1) {
                    ((__bf16*)Dv)[idx] = f2bf(v);
                } else if (op == 2) {
                    ((float*)Dv)[idx] = 1.f / (1.f + __expf(-v));
                } else if (op == 3) {
                    float s = v / (1.f + __expf(-v));      // silu
                    ((__bf16*)Dv)[idx] = f2bf(s * aux[idx]);
                } else {
                    ((float*)Dv)[idx] = v + aux[idx];
                }
            }
        }
    }
}

// ---------------- sliding-window causal attention (flash style) -------------
// grid = B * H * (T/128); block = 256 (8 waves, each wave = 16 query rows)
__global__ __launch_bounds__(256) void attn_kernel(
    const __bf16* __restrict__ qkv,   // [B*T][3072]  q|k|v
    const __bf16* __restrict__ vT,    // [B*H*64][TPAD]
    const float*  __restrict__ gate,  // [B*T][1024]  sigmoid already applied
    __bf16* __restrict__ outg) {      // [B*T][1024]  gated attention output
    __shared__ __align__(16) __bf16 sP[8][16 * 40];
    const int tid  = threadIdx.x;
    const int lane = tid & 31;
    const int wave = tid >> 5;
    const int lr = lane & 15;
    const int hi = lane >> 4;

    const int qblk = blockIdx.x % (TT / 128);
    const int bh   = blockIdx.x / (TT / 128);
    const int h = bh % N_HEADS, b = bh / N_HEADS;
    const int q0 = qblk * 128 + wave * 16;
    const float scale = 0.125f;   // 1/sqrt(64)

    // Q fragments: two K=32 chunks over d_head=64
    v16bf qf[2];
    {
        int t = q0 + lr;
        const __bf16* qrow = qkv + (size_t)(b * TT + t) * 3072 + h * D_HEAD;
        int db = hi ? 8 : 0;
        #pragma unroll
        for (int c = 0; c < 2; ++c) {
            const __bf16* p = qrow + c * 32 + db;
            qf[c] = make_frag(*(const uint4*)p, *(const uint4*)(p + 16));
        }
    }

    v8f o[4] = {};
    float mrow[8], lrow[8];
    #pragma unroll
    for (int r = 0; r < 8; ++r) { mrow[r] = -3e38f; lrow[r] = 0.f; }

    int lo = q0 - (WINDOW - 1); if (lo < 0) lo = 0;
    for (int kt = lo & ~31; kt <= q0 + 15; kt += 32) {
        // ---- scores: two 16-key subtiles, contraction over d (2 x K=32) ----
        v8f s[2] = {};
        #pragma unroll
        for (int sub = 0; sub < 2; ++sub) {
            int t = kt + sub * 16 + lr;
            if (t > TT - 1) t = TT - 1;                 // masked anyway
            const __bf16* krow =
                qkv + (size_t)(b * TT + t) * 3072 + 1024 + h * D_HEAD;
            #pragma unroll
            for (int c = 0; c < 2; ++c) {
                const __bf16* p = krow + c * 32 + (hi ? 8 : 0);
                v16bf kf = make_frag(*(const uint4*)p, *(const uint4*)(p + 16));
                s[sub] = wmma_bf16(qf[c], kf, s[sub]);
            }
        }
        // ---- mask + online softmax (row stats across 16-lane half-wave) ----
        float p0[8], p1[8], alpha[8];
        #pragma unroll
        for (int r = 0; r < 8; ++r) {
            int qr  = q0 + r + hi * 8;
            int k0i = kt + lr;
            int k1i = kt + 16 + lr;
            float s0 = s[0][r] * scale;
            float s1 = s[1][r] * scale;
            bool ok0 = (k0i <= qr) && (qr - k0i < WINDOW);
            bool ok1 = (k1i <= qr) && (qr - k1i < WINDOW);
            s0 = ok0 ? s0 : -3e38f;
            s1 = ok1 ? s1 : -3e38f;
            float mx = fmaxf(s0, s1);
            #pragma unroll
            for (int m = 1; m < 16; m <<= 1) mx = fmaxf(mx, __shfl_xor(mx, m, 32));
            float nm = fmaxf(mrow[r], mx);
            float a  = __expf(mrow[r] - nm);
            float e0 = __expf(s0 - nm);
            float e1 = __expf(s1 - nm);
            float rs = e0 + e1;
            #pragma unroll
            for (int m = 1; m < 16; m <<= 1) rs += __shfl_xor(rs, m, 32);
            mrow[r] = nm;
            lrow[r] = lrow[r] * a + rs;
            alpha[r] = a;
            p0[r] = e0; p1[r] = e1;
        }
        #pragma unroll
        for (int dt = 0; dt < 4; ++dt)
            #pragma unroll
            for (int r = 0; r < 8; ++r) o[dt][r] *= alpha[r];

        // ---- C-layout P -> A-layout via wave-private LDS ----
        __bf16* P = sP[wave];
        #pragma unroll
        for (int r = 0; r < 8; ++r) {
            int row = r + hi * 8;
            P[row * 40 + lr]      = f2bf(p0[r]);
            P[row * 40 + 16 + lr] = f2bf(p1[r]);
        }
        asm volatile("s_wait_dscnt 0" ::: "memory");
        v16bf pf;
        {
            const __bf16* p = P + lr * 40 + (hi ? 8 : 0);
            pf = make_frag(*(const uint4*)p, *(const uint4*)(p + 16));
        }
        // ---- O += P @ V (contraction over 32 keys, 4 d-tiles) ----
        const __bf16* vbase = vT + ((size_t)(b * N_HEADS + h) * D_HEAD) * TPAD;
        #pragma unroll
        for (int dt = 0; dt < 4; ++dt) {
            const __bf16* p =
                vbase + (size_t)(dt * 16 + lr) * TPAD + kt + (hi ? 8 : 0);
            v16bf vf = make_frag(*(const uint4*)p, *(const uint4*)(p + 16));
            o[dt] = wmma_bf16(pf, vf, o[dt]);
        }
    }

    // ---- epilogue: O/l * gate, store bf16 ----
    #pragma unroll
    for (int dt = 0; dt < 4; ++dt) {
        #pragma unroll
        for (int r = 0; r < 8; ++r) {
            int qr = q0 + r + hi * 8;
            int d  = dt * 16 + lr;
            size_t gi = (size_t)(b * TT + qr) * D_MODEL + h * D_HEAD + d;
            float v = o[dt][r] / lrow[r];
            outg[gi] = f2bf(v * gate[gi]);
        }
    }
}

// ---------------- host orchestration ----------------
extern "C" void kernel_launch(void* const* d_in, const int* in_sizes, int n_in,
                              void* d_out, int out_size, void* d_ws, size_t ws_size,
                              hipStream_t stream) {
    (void)in_sizes; (void)n_in; (void)out_size; (void)ws_size;
    const float* x    = (const float*)d_in[0];
    const float* ln1  = (const float*)d_in[1];
    const float* qkvw = (const float*)d_in[2];
    const float* gatw = (const float*)d_in[3];
    const float* outw = (const float*)d_in[4];
    const float* ln2  = (const float*)d_in[5];
    const float* wg   = (const float*)d_in[6];
    const float* wu   = (const float*)d_in[7];
    const float* wo   = (const float*)d_in[8];
    float* out = (float*)d_out;

    char* ws = (char*)d_ws;
    size_t off = 0;
    auto alloc = [&](size_t bytes) -> char* {
        char* p = ws + off;
        off += (bytes + 255) & ~(size_t)255;
        return p;
    };
    __bf16* wqkv_bf = (__bf16*)alloc((size_t)3072 * 1024 * 2);
    __bf16* wgat_bf = (__bf16*)alloc((size_t)1024 * 1024 * 2);
    __bf16* wout_bf = (__bf16*)alloc((size_t)1024 * 1024 * 2);
    __bf16* wg_bf   = (__bf16*)alloc((size_t)4096 * 1024 * 2);
    __bf16* wu_bf   = (__bf16*)alloc((size_t)4096 * 1024 * 2);
    __bf16* wo_bf   = (__bf16*)alloc((size_t)1024 * 4096 * 2);
    __bf16* h1_bf   = (__bf16*)alloc((size_t)MROWS * 1024 * 2);
    __bf16* qkv_bf  = (__bf16*)alloc((size_t)MROWS * 3072 * 2);
    __bf16* vT_bf   = (__bf16*)alloc((size_t)BB * N_HEADS * D_HEAD * TPAD * 2);
    float*  gate_f  = (float*) alloc((size_t)MROWS * 1024 * 4);
    __bf16* attg_bf = (__bf16*)alloc((size_t)MROWS * 1024 * 2);
    float*  x1_f    = (float*) alloc((size_t)MROWS * 1024 * 4);
    __bf16* h2_bf   = (__bf16*)alloc((size_t)MROWS * 1024 * 2);
    float*  a1_f    = (float*) alloc((size_t)MROWS * 4096 * 4);
    __bf16* gg_bf   = (__bf16*)alloc((size_t)MROWS * 4096 * 2);

    // weight conversions fp32 -> bf16
    cvt_bf16_kernel<<<2048, 256, 0, stream>>>(qkvw, wqkv_bf, (size_t)3072 * 1024);
    cvt_bf16_kernel<<<1024, 256, 0, stream>>>(gatw, wgat_bf, (size_t)1024 * 1024);
    cvt_bf16_kernel<<<1024, 256, 0, stream>>>(outw, wout_bf, (size_t)1024 * 1024);
    cvt_bf16_kernel<<<2048, 256, 0, stream>>>(wg,   wg_bf,   (size_t)4096 * 1024);
    cvt_bf16_kernel<<<2048, 256, 0, stream>>>(wu,   wu_bf,   (size_t)4096 * 1024);
    cvt_bf16_kernel<<<2048, 256, 0, stream>>>(wo,   wo_bf,   (size_t)1024 * 4096);

    // h1 = rmsnorm(x, ln1)
    rmsnorm_kernel<<<MROWS, 256, 0, stream>>>(x, ln1, h1_bf);

    // qkv = h1 @ qkv_w^T   (bf16 out)
    gemm_bf16_kernel<<<dim3(3072 / 128, MROWS / 128), 256, 0, stream>>>(
        h1_bf, 1024, wqkv_bf, 1024, 1024, qkv_bf, 3072, 1, nullptr);
    // gate = sigmoid(h1 @ gate_w^T)   (f32 out)
    gemm_bf16_kernel<<<dim3(1024 / 128, MROWS / 128), 256, 0, stream>>>(
        h1_bf, 1024, wgat_bf, 1024, 1024, gate_f, 1024, 2, nullptr);

    // vT = transpose(v) with zero tail pad
    vtrans_kernel<<<2048, 256, 0, stream>>>(qkv_bf, vT_bf);

    // gated windowed attention
    attn_kernel<<<BB * N_HEADS * (TT / 128), 256, 0, stream>>>(
        qkv_bf, vT_bf, gate_f, attg_bf);

    // x1 = x + attg @ out_w^T
    gemm_bf16_kernel<<<dim3(1024 / 128, MROWS / 128), 256, 0, stream>>>(
        attg_bf, 1024, wout_bf, 1024, 1024, x1_f, 1024, 4, x);

    // h2 = rmsnorm(x1, ln2)
    rmsnorm_kernel<<<MROWS, 256, 0, stream>>>(x1_f, ln2, h2_bf);

    // a1 = h2 @ wu^T (f32)
    gemm_bf16_kernel<<<dim3(4096 / 128, MROWS / 128), 256, 0, stream>>>(
        h2_bf, 1024, wu_bf, 1024, 1024, a1_f, 4096, 0, nullptr);
    // gg = silu(h2 @ wg^T) * a1 (bf16)
    gemm_bf16_kernel<<<dim3(4096 / 128, MROWS / 128), 256, 0, stream>>>(
        h2_bf, 1024, wg_bf, 1024, 1024, gg_bf, 4096, 3, a1_f);
    // out = x1 + gg @ wo^T
    gemm_bf16_kernel<<<dim3(1024 / 128, MROWS / 128), 256, 0, stream>>>(
        gg_bf, 4096, wo_bf, 4096, 4096, out, 1024, 4, x1_f);
}
